// MultiHeadAttn_86663850099105
// MI455X (gfx1250) — matrix operand order repaired and checked by reading, not compiled, not run on hardware
//
#include <hip/hip_runtime.h>
#include <hip/hip_bf16.h>

// ---------------------------------------------------------------------------
// MultiHeadAttn block for MI455X (gfx1250, wave32, WMMA, TDM).
// bf16 WMMA f32-accumulate; flash-attention fusion; Tensor Data Mover stages
// K tiles / GEMM A tiles into LDS (hardware padding via D# pad fields); DPP
// butterflies for softmax row reductions (no ds_bpermute).
// ---------------------------------------------------------------------------

typedef __bf16 bf16;
typedef bf16  v16bf __attribute__((ext_vector_type(16)));
typedef float v8f   __attribute__((ext_vector_type(8)));
typedef unsigned int v4u  __attribute__((ext_vector_type(4)));
typedef unsigned int v8u  __attribute__((ext_vector_type(8)));

#define B_    4
#define S_    2048
#define DM_   1024
#define NH_   16
#define HD_   64
#define SCALE_ 0.125f   /* 1/sqrt(64) */

#define WMMA_BF16(a, b, c) \
  __builtin_amdgcn_wmma_f32_16x16x32_bf16(false, (a), false, (b), (short)0, (c), false, false)

// --- WMMA fragment loads from LDS ------------------------------------------
// A (16x32 bf16): lanes 0-15 row M=lane, K {0..7}U{16..23}; lanes 16-31 same
// rows, K {8..15}U{24..31}.
static __device__ __forceinline__ v16bf frag_a(const bf16* base, int row, int stride,
                                               int hf, int kOff) {
  union { v16bf v; v4u u[2]; } r;
  const bf16* p = base + row * stride + kOff + hf * 8;
  r.u[0] = *(const v4u*)p;
  r.u[1] = *(const v4u*)(p + 16);
  return r.v;
}
// B (32x16 bf16) stored transposed in LDS as [n][k]: lanes 0-15 col N=lane,
// K {0..7}U{16..23}; lanes 16-31 K {8..15}U{24..31}.
static __device__ __forceinline__ v16bf frag_b(const bf16* base, int row, int stride,
                                               int hf, int kOff) {
  union { v16bf v; v4u u[2]; } r;
  const bf16* p = base + row * stride + kOff + hf * 8;
  r.u[0] = *(const v4u*)p;
  r.u[1] = *(const v4u*)(p + 16);
  return r.v;
}

// --- DPP 16-lane butterfly reductions (VALU only, no LDS) ------------------
template <int CTRL>
static __device__ __forceinline__ float dppmv(float v) {
  return __int_as_float(
      __builtin_amdgcn_update_dpp(0, __float_as_int(v), CTRL, 0xF, 0xF, true));
}
static __device__ __forceinline__ float red16_max(float x) {
  x = fmaxf(x, dppmv<0xB1>(x));   // quad_perm(1,0,3,2)  : xor1
  x = fmaxf(x, dppmv<0x4E>(x));   // quad_perm(2,3,0,1)  : xor2
  x = fmaxf(x, dppmv<0x141>(x));  // row_half_mirror     : pairs groups of 4
  x = fmaxf(x, dppmv<0x140>(x));  // row_mirror          : pairs groups of 8
  return x;
}
static __device__ __forceinline__ float red16_sum(float x) {
  x += dppmv<0xB1>(x);
  x += dppmv<0x4E>(x);
  x += dppmv<0x141>(x);
  x += dppmv<0x140>(x);
  return x;
}

// --- Tensor Data Mover: 2-D bf16 tile -> LDS (with HW padding) -------------
// D# group0/group1 per CDNA5 ISA ch.8 (2-byte elements, 2-D tile, no cluster).
static __device__ __forceinline__ void tdm_load_2d(
    unsigned ldsAddr, const void* gptr,
    unsigned tileD0, unsigned tileD1,          // elements, rows
    unsigned tensD0, unsigned tensD1,          // elements, rows
    unsigned long long strideElems,            // row stride, elements
    unsigned padIntervalCode, unsigned padAmountCode) {
  unsigned long long ga = (unsigned long long)(size_t)gptr;
  v4u g0;
  g0[0] = 1u;                                        // count=1, user descriptor
  g0[1] = ldsAddr;                                   // lds_addr (bytes)
  g0[2] = (unsigned)(ga & 0xFFFFFFFFu);              // global_addr lo
  g0[3] = (unsigned)((ga >> 32) & 0x1FFFFFFu)        // global_addr hi (57b)
          | 0x80000000u;                             // type=2 ("image")
  v8u g1;
  g1[0] = (1u << 16)                                 // data_size = 2 bytes
          | (1u << 20)                               // pad_enable
          | (padIntervalCode << 22)
          | (padAmountCode << 25);
  g1[1] = (tensD0 & 0xFFFFu) << 16;                  // tensor_dim0[15:0]
  g1[2] = (tensD0 >> 16) | ((tensD1 & 0xFFFFu) << 16);
  g1[3] = (tensD1 >> 16) | (tileD0 << 16);           // tile_dim0
  g1[4] = tileD1;                                    // tile_dim1 (tile_dim2=0)
  g1[5] = (unsigned)(strideElems & 0xFFFFFFFFull);   // tensor_dim0_stride lo
  g1[6] = (unsigned)((strideElems >> 32) & 0xFFFFull);
  g1[7] = 0u;
  asm volatile("tensor_load_to_lds %0, %1" :: "s"(g0), "s"(g1) : "memory");
}

// ------------------------- fp32 -> bf16 convert ----------------------------
__global__ __launch_bounds__(256) void f32_to_bf16_kernel(const float* __restrict__ in,
                                                          bf16* __restrict__ out, int n) {
  int i = blockIdx.x * 256 + threadIdx.x;
  if (i < n) out[i] = (bf16)in[i];
}

// ------------------------- tiled bf16 WMMA GEMM ----------------------------
// C[M,N] = A[M,K] * B[K,N] (+bias).  128x128 block tile, 8 waves (4x2),
// each wave 32x64 = 2x4 grid of 16x16x32 WMMAs.  K-step = 32.
// A tile staged by TDM (HW-padded stride 40); B tile staged manually with
// transpose into [n][k].
#define LSTR 40   // padded LDS row stride (elements); 80 B, 16-B aligned rows

template <typename OutT, bool HAS_BIAS>
__global__ __launch_bounds__(256) void gemm_bf16_kernel(
    const bf16* __restrict__ A, const bf16* __restrict__ Bm,
    const float* __restrict__ bias, OutT* __restrict__ C,
    int M, int N, int K) {
  __shared__ alignas(16) bf16 As[128 * LSTR];
  __shared__ alignas(16) bf16 Bs[128 * LSTR];

  const int tid   = threadIdx.x;
  const int lane  = tid & 31;
  const int wave  = tid >> 5;
  const int waveM = wave & 3;
  const int waveN = wave >> 2;
  const int hf    = lane >> 4;
  const int l15   = lane & 15;
  const int blockM = blockIdx.y * 128;
  const int blockN = blockIdx.x * 128;
  const unsigned AsAddr = (unsigned)(size_t)(void*)As;

  v8f acc[2][4] = {};

  for (int k0 = 0; k0 < K; k0 += 32) {
    __syncthreads();
    if (wave == 0) {
      // TDM: A tile 128 rows x 32 cols, row stride K, pad 16->+4 DWORDs
      tdm_load_2d(AsAddr, A + (size_t)blockM * K + k0,
                  /*tile*/ 32, 128, /*tensor*/ 32, 128,
                  (unsigned long long)K, /*interval 16 dw*/ 3, /*pad 4 dw*/ 3);
    }
    {  // stage B tile 32x128, transposed into LDS as [n][k]
      int kr = tid >> 3;
      int nc = (tid & 7) * 16;
      const bf16* g = Bm + (size_t)(k0 + kr) * N + blockN + nc;
      union { v16bf v; v4u u[2]; } t;
      t.u[0] = *(const v4u*)g;
      t.u[1] = *(const v4u*)(g + 8);
#pragma unroll
      for (int i = 0; i < 16; ++i) Bs[(nc + i) * LSTR + kr] = t.v[i];
    }
    if (wave == 0) __builtin_amdgcn_s_wait_tensorcnt(0);
    __syncthreads();

    v16bf af[2], bfr[4];
#pragma unroll
    for (int mt = 0; mt < 2; ++mt)
      af[mt] = frag_a(As, waveM * 32 + mt * 16 + l15, LSTR, hf, 0);
#pragma unroll
    for (int nt = 0; nt < 4; ++nt)
      bfr[nt] = frag_b(Bs, waveN * 64 + nt * 16 + l15, LSTR, hf, 0);
#pragma unroll
    for (int mt = 0; mt < 2; ++mt)
#pragma unroll
      for (int nt = 0; nt < 4; ++nt)
        acc[mt][nt] = WMMA_BF16(af[mt], bfr[nt], acc[mt][nt]);
  }

#pragma unroll
  for (int mt = 0; mt < 2; ++mt)
#pragma unroll
    for (int nt = 0; nt < 4; ++nt) {
      int col = blockN + waveN * 64 + nt * 16 + l15;
      float bv = HAS_BIAS ? bias[col] : 0.0f;
#pragma unroll
      for (int r = 0; r < 8; ++r) {
        int row = blockM + waveM * 32 + mt * 16 + hf * 8 + r;
        float v = acc[mt][nt][r] + bv;
        C[(size_t)row * N + col] = (OutT)v;
      }
    }
}

// ------------------------- flash attention ---------------------------------
// grid.x = S/128 (query chunks), grid.y = B*H.  8 waves/block, each wave a
// 16-row query tile; 64-key blocks streamed through LDS (K via TDM, V via
// manual transpose); online softmax with DPP reductions; 16 WMMAs/iteration.
#define FSTR 72   // LDS row stride (144 B, 16-B aligned): 64 + 8 pad elements

__global__ __launch_bounds__(256) void flash_attn_kernel(
    const bf16* __restrict__ qkv,           // [B*S, 3072] bf16 (Q|K|V)
    const unsigned char* __restrict__ mask, // [B, S] bool
    bf16* __restrict__ attnvec) {           // [B*S, 1024] bf16
  __shared__ alignas(16) bf16 Ks[64 * FSTR];      // K block [key][hd]
  __shared__ alignas(16) bf16 Vt[64 * FSTR];      // V block transposed [hd][key]
  __shared__ alignas(16) bf16 Plds[8][16 * FSTR]; // per-wave P tile [16][64]

  const int tid  = threadIdx.x;
  const int lane = tid & 31;
  const int w    = tid >> 5;
  const int hf   = lane >> 4;
  const int l15  = lane & 15;
  const int b  = blockIdx.y >> 4;
  const int h  = blockIdx.y & 15;
  const int q0 = blockIdx.x * 128 + w * 16;
  const unsigned KsAddr = (unsigned)(size_t)(void*)Ks;

  const float NEG_INF = -__builtin_inff();

  // Q fragments (A layout), head-dim slices 0..31 / 32..63
  const bf16* qrow = qkv + (size_t)(b * S_ + q0 + l15) * 3072 + h * HD_;
  v16bf qf[2];
  {
    union { v16bf v; v4u u[2]; } t;
    t.u[0] = *(const v4u*)(qrow + hf * 8);
    t.u[1] = *(const v4u*)(qrow + 16 + hf * 8);
    qf[0] = t.v;
    t.u[0] = *(const v4u*)(qrow + 32 + hf * 8);
    t.u[1] = *(const v4u*)(qrow + 48 + hf * 8);
    qf[1] = t.v;
  }

  v8f o[4] = {};
  float rowMax[8], rowSum[8];
#pragma unroll
  for (int r = 0; r < 8; ++r) { rowMax[r] = NEG_INF; rowSum[r] = 0.0f; }

  for (int kb = 0; kb < S_ / 64; ++kb) {
    if (w == 0) {
      // TDM: K block 64 keys x 64 hd, row stride 3072, pad 32 dw -> +4 dw
      tdm_load_2d(KsAddr,
                  qkv + (size_t)(b * S_ + kb * 64) * 3072 + DM_ + h * HD_,
                  /*tile*/ 64, 64, /*tensor*/ 64, 64, 3072ull,
                  /*interval 32 dw*/ 4, /*pad 4 dw*/ 3);
    }
    {  // stage V block transposed [hd][key]; 2 chunks of 8 elements per thread
#pragma unroll
      for (int c2 = 0; c2 < 2; ++c2) {
        int c  = tid * 2 + c2;
        int j  = c >> 3;             // key row 0..63
        int cc = (c & 7) * 8;        // hd chunk
        const bf16* g =
            qkv + (size_t)(b * S_ + kb * 64 + j) * 3072 + 2 * DM_ + h * HD_ + cc;
        union { v16bf v; v4u u[2]; } t;
        t.u[0] = *(const v4u*)g;
#pragma unroll
        for (int i = 0; i < 8; ++i) Vt[(cc + i) * FSTR + j] = t.v[i];
      }
    }
    if (w == 0) __builtin_amdgcn_s_wait_tensorcnt(0);
    __syncthreads();

    // scores: 16 queries x 64 keys = four 16x16 tiles, 2 head-dim slices
    v8f s[4] = {};
#pragma unroll
    for (int nt = 0; nt < 4; ++nt)
#pragma unroll
      for (int sl = 0; sl < 2; ++sl) {
        v16bf kfr = frag_b(Ks, nt * 16 + l15, FSTR, hf, sl * 32);
        s[nt] = WMMA_BF16(qf[sl], kfr, s[nt]);
      }

    // scale + key mask (lane's column == one key per tile)
#pragma unroll
    for (int nt = 0; nt < 4; ++nt) {
      int key = kb * 64 + nt * 16 + l15;
      bool ok = mask[b * S_ + key] != 0;
#pragma unroll
      for (int r = 0; r < 8; ++r)
        s[nt][r] = ok ? s[nt][r] * SCALE_ : NEG_INF;
    }

    // online softmax row stats (DPP butterfly across the 16 lanes of a half)
    float corr[8];
#pragma unroll
    for (int r = 0; r < 8; ++r) {
      float mx = fmaxf(fmaxf(s[0][r], s[1][r]), fmaxf(s[2][r], s[3][r]));
      mx = red16_max(mx);
      float nm = fmaxf(rowMax[r], mx);
      float ls = 0.0f;
#pragma unroll
      for (int nt = 0; nt < 4; ++nt) {
        float p = __expf(s[nt][r] - nm);
        s[nt][r] = p;
        ls += p;
      }
      ls = red16_sum(ls);
      float c = __expf(rowMax[r] - nm);
      rowSum[r] = rowSum[r] * c + ls;
      rowMax[r] = nm;
      corr[r] = c;
    }
#pragma unroll
    for (int t = 0; t < 4; ++t)
#pragma unroll
      for (int r = 0; r < 8; ++r) o[t][r] *= corr[r];

    // P (C layout) -> LDS -> A-fragment layout for PV
    bf16* Pw = &Plds[w][0];
#pragma unroll
    for (int nt = 0; nt < 4; ++nt)
#pragma unroll
      for (int r = 0; r < 8; ++r)
        Pw[(hf * 8 + r) * FSTR + nt * 16 + l15] = (bf16)s[nt][r];
    __syncthreads();

    v16bf pf[2];
    pf[0] = frag_a(Pw, l15, FSTR, hf, 0);    // keys 0..31
    pf[1] = frag_a(Pw, l15, FSTR, hf, 32);   // keys 32..63
#pragma unroll
    for (int ct = 0; ct < 4; ++ct) {
      v16bf v0 = frag_b(Vt, ct * 16 + l15, FSTR, hf, 0);
      v16bf v1 = frag_b(Vt, ct * 16 + l15, FSTR, hf, 32);
      o[ct] = WMMA_BF16(pf[0], v0, o[ct]);
      o[ct] = WMMA_BF16(pf[1], v1, o[ct]);
    }
    __syncthreads();   // protect Ks/Vt before next iteration's staging
  }

  // normalize and write attn_vec (bf16)
#pragma unroll
  for (int ct = 0; ct < 4; ++ct)
#pragma unroll
    for (int r = 0; r < 8; ++r) {
      int row = b * S_ + q0 + hf * 8 + r;
      int col = h * HD_ + ct * 16 + l15;
      attnvec[(size_t)row * DM_ + col] = (bf16)(o[ct][r] / rowSum[r]);
    }
}

// ---------------------- residual + LayerNorm -------------------------------
__global__ __launch_bounds__(256) void resid_ln_kernel(
    const float* __restrict__ x, const float* __restrict__ attn_out,
    const float* __restrict__ gamma, const float* __restrict__ beta,
    float* __restrict__ out) {
  const int row = blockIdx.x;
  const int tid = threadIdx.x;
  __shared__ float redS[8], redQ[8];
  __shared__ float s_mu, s_rstd;

  float v[4], s = 0.0f, q = 0.0f;
#pragma unroll
  for (int i = 0; i < 4; ++i) {
    int idx = tid + i * 256;
    float t = x[(size_t)row * DM_ + idx] + attn_out[(size_t)row * DM_ + idx];
    v[i] = t; s += t; q += t * t;
  }
#pragma unroll
  for (int d = 1; d < 32; d <<= 1) { s += __shfl_xor(s, d, 32); q += __shfl_xor(q, d, 32); }
  if ((tid & 31) == 0) { redS[tid >> 5] = s; redQ[tid >> 5] = q; }
  __syncthreads();
  if (tid == 0) {
    float S = 0.0f, Q = 0.0f;
    for (int i = 0; i < 8; ++i) { S += redS[i]; Q += redQ[i]; }
    float mu = S * (1.0f / DM_);
    float var = Q * (1.0f / DM_) - mu * mu;
    s_mu = mu;
    s_rstd = rsqrtf(var + 1e-3f);
  }
  __syncthreads();
  float mu = s_mu, rstd = s_rstd;
#pragma unroll
  for (int i = 0; i < 4; ++i) {
    int idx = tid + i * 256;
    out[(size_t)row * DM_ + idx] = (v[i] - mu) * rstd * gamma[idx] + beta[idx];
  }
}

// ---------------------------------------------------------------------------
extern "C" void kernel_launch(void* const* d_in, const int* in_sizes, int n_in,
                              void* d_out, int out_size, void* d_ws, size_t ws_size,
                              hipStream_t stream) {
  const float*         inputs = (const float*)d_in[0];
  const unsigned char* amask  = (const unsigned char*)d_in[1];
  const float*         W_qkv  = (const float*)d_in[2];
  const float*         b_qkv  = (const float*)d_in[3];
  const float*         W_o    = (const float*)d_in[4];
  const float*         gamma  = (const float*)d_in[5];
  const float*         beta   = (const float*)d_in[6];
  float*               out    = (float*)d_out;

  const size_t BS = (size_t)B_ * S_;          // 8192 rows
  char* ws = (char*)d_ws;
  auto carve = [&](size_t bytes) {
    char* p = ws;
    ws += (bytes + 255) & ~(size_t)255;
    return p;
  };
  bf16*  Xb       = (bf16*)carve(BS * DM_ * sizeof(bf16));
  bf16*  Wqkvb    = (bf16*)carve((size_t)DM_ * 3 * DM_ * sizeof(bf16));
  bf16*  Wob      = (bf16*)carve((size_t)DM_ * DM_ * sizeof(bf16));
  bf16*  qkvb     = (bf16*)carve(BS * 3 * DM_ * sizeof(bf16));
  bf16*  attnvec  = (bf16*)carve(BS * DM_ * sizeof(bf16));
  float* attn_out = (float*)carve(BS * DM_ * sizeof(float));

  {
    int n = (int)(BS * DM_);
    f32_to_bf16_kernel<<<(n + 255) / 256, 256, 0, stream>>>(inputs, Xb, n);
  }
  {
    int n = DM_ * 3 * DM_;
    f32_to_bf16_kernel<<<(n + 255) / 256, 256, 0, stream>>>(W_qkv, Wqkvb, n);
  }
  {
    int n = DM_ * DM_;
    f32_to_bf16_kernel<<<(n + 255) / 256, 256, 0, stream>>>(W_o, Wob, n);
  }

  // QKV projection: [8192,1024] x [1024,3072] + bias -> bf16
  gemm_bf16_kernel<bf16, true><<<dim3(3 * DM_ / 128, (int)(BS / 128)), 256, 0, stream>>>(
      Xb, Wqkvb, b_qkv, qkvb, (int)BS, 3 * DM_, DM_);

  // fused flash attention -> attn_vec bf16
  flash_attn_kernel<<<dim3(S_ / 128, B_ * NH_), 256, 0, stream>>>(qkvb, amask, attnvec);

  // output projection: [8192,1024] x [1024,1024] -> fp32
  gemm_bf16_kernel<float, false><<<dim3(DM_ / 128, (int)(BS / 128)), 256, 0, stream>>>(
      attnvec, Wob, nullptr, attn_out, (int)BS, DM_, DM_);

  // residual + LayerNorm -> d_out
  resid_ln_kernel<<<(int)BS, 256, 0, stream>>>(inputs, attn_out, gamma, beta, out);
}
